// DiscreteMamba2_33045478375466
// MI455X (gfx1250) — matrix operand order repaired and checked by reading, not compiled, not run on hardware
//
#include <hip/hip_runtime.h>
#include <hip/hip_bf16.h>
#include <math.h>

#define BATCH   2
#define LSEQ    4096
#define DMODEL  1024
#define DINNER  2048
#define DSTATE  64
#define NHEADS  32
#define HEADDIM 64           // DINNER / NHEADS
#define CONVK   4
#define CHUNK   128
#define NCHUNK  (LSEQ / CHUNK)                 // 32
#define DXBC    (DINNER + 2 * NHEADS * DSTATE) // 6144
#define DPROJ   (DXBC + DINNER + NHEADS)       // 8224

#define USE_ASYNC_LDS 1

typedef __bf16 bf16_t;
typedef __attribute__((ext_vector_type(16))) __bf16 v16bf;
typedef __attribute__((ext_vector_type(8)))  float  v8f;

union FragAB { v16bf v; bf16_t e[16]; uint4 q[2]; };
union FragCD { v8f   v; float  e[8];  };

// CDNA5 ISA 16-bit A-matrix 16x32 layout: lane half h (=lane/16),
// element e -> K:  e<8: K = h*8+e ;  e>=8: K = 16 + h*8 + (e-8)
// => two contiguous 8-element (16B) runs at [h*8] and [16 + h*8].
__device__ __forceinline__ int ka_idx(int e, int h) {
  return e + 8 * h + (e >= 8 ? 8 : 0);
}
// Contiguous k order, used here for both A and B: lane half h: K = 16*h + e (contiguous 16).
__device__ __forceinline__ int kb_idx(int e, int h) { return 16 * h + e; }

__device__ __forceinline__ v8f wmma_bf16(v16bf a, v16bf b, v8f c) {
  return __builtin_amdgcn_wmma_f32_16x16x32_bf16(
      false, a, false, b, (short)0, c, false, false);
}

__device__ __forceinline__ float sigmoidf(float x) {
  return 1.0f / (1.0f + __expf(-x));
}

// gfx1250 async global->LDS copy (ASYNCcnt-tracked DMA into LDS).
__device__ __forceinline__ void async_load_lds_b128(unsigned lds_byte_off,
                                                    const void* gptr) {
#if USE_ASYNC_LDS
  asm volatile("global_load_async_to_lds_b128 %0, %1, off"
               :: "v"(lds_byte_off), "v"(gptr) : "memory");
#endif
}
__device__ __forceinline__ void wait_async_lds() {
#if USE_ASYNC_LDS
  asm volatile("s_wait_asynccnt 0x0" ::: "memory");
#endif
}
__device__ __forceinline__ unsigned lds_off32(const void* p) {
  // generic pointers to LDS carry the LDS byte address in the low 32 bits
  return (unsigned)(unsigned long long)p;
}

// ---------------------------------------------------------------------------
// C[m,n] = sum_k A[m,k]*Bw[n,k], operands pre-split into bf16 hi/lo arrays.
// 3 WMMAs per 16x16x32 step (hh + hl + lh) ~ fp32 accuracy.
// Block: 256 thr (8 waves); tile 64(M) x 128(N, 1 n-tile/wave); K panel 128.
// A panel staged to LDS via async global->LDS b128 copies (pure DMA, no cvt).
// N-edge handled by clamping (duplicate-compute of last tile, identical
// values re-stored) so there is NO exec-mask divergence anywhere.
// ---------------------------------------------------------------------------
__global__ __launch_bounds__(256) void gemm_nt_bf16x2(
    const bf16_t* __restrict__ Ahi, const bf16_t* __restrict__ Alo,
    const bf16_t* __restrict__ Bhi, const bf16_t* __restrict__ Blo,
    float* __restrict__ C, int M, int N, int K) {
  __shared__ bf16_t Ahs[64][128];  // 16 KB
  __shared__ bf16_t Als[64][128];  // 16 KB

  const int wave  = threadIdx.x >> 5;
  const int lane  = threadIdx.x & 31;
  const int lrow  = lane & 15;
  const int lhalf = lane >> 4;
  const int m0    = blockIdx.x * 64;
  const int NT    = N >> 4;
  int ntile = blockIdx.y * 8 + wave;
  if (ntile >= NT) ntile = NT - 1;  // benign duplicate of last tile
  const int n0 = ntile * 16;

  FragCD acc[4];
#pragma unroll
  for (int t = 0; t < 4; ++t)
#pragma unroll
    for (int i = 0; i < 8; ++i) acc[t].e[i] = 0.0f;

  for (int k0 = 0; k0 < K; k0 += 128) {
    __syncthreads();
    // Stage 64x128 bf16 hi/lo panels: 1024 16B chunks each, 256 threads.
    for (int i = threadIdx.x; i < 64 * 16; i += 256) {
      int r = i >> 4, ch = i & 15;  // ch indexes 8-element (16B) chunks
      const bf16_t* gh = &Ahi[(size_t)(m0 + r) * K + k0 + ch * 8];
      const bf16_t* gl = &Alo[(size_t)(m0 + r) * K + k0 + ch * 8];
#if USE_ASYNC_LDS
      async_load_lds_b128(lds_off32(&Ahs[r][ch * 8]), gh);
      async_load_lds_b128(lds_off32(&Als[r][ch * 8]), gl);
#else
      *(uint4*)&Ahs[r][ch * 8] = *(const uint4*)gh;
      *(uint4*)&Als[r][ch * 8] = *(const uint4*)gl;
#endif
    }
    if (k0 + 128 < K)  // gfx1250 prefetch hint for next panel
      __builtin_prefetch(&Ahi[(size_t)m0 * K + k0 + 128], 0, 1);
    wait_async_lds();
    __syncthreads();

#pragma unroll
    for (int kk = 0; kk < 128; kk += 32) {
      // ---- preload ALL fragments for this k-step, then burst 12 WMMAs ----
      FragAB bhf, blf;
      {
        const size_t boff = (size_t)(n0 + lrow) * K + k0 + kk + 8 * lhalf;
        bhf.q[0] = ((const uint4*)&Bhi[boff])[0];
        bhf.q[1] = ((const uint4*)&Bhi[boff])[2];
        blf.q[0] = ((const uint4*)&Blo[boff])[0];
        blf.q[1] = ((const uint4*)&Blo[boff])[2];
      }
      FragAB ah[4], al[4];
#pragma unroll
      for (int t = 0; t < 4; ++t) {
        const bf16_t* arh = &Ahs[t * 16 + lrow][kk + 8 * lhalf];
        const bf16_t* arl = &Als[t * 16 + lrow][kk + 8 * lhalf];
        ah[t].q[0] = *(const uint4*)arh;
        ah[t].q[1] = *(const uint4*)(arh + 16);
        al[t].q[0] = *(const uint4*)arl;
        al[t].q[1] = *(const uint4*)(arl + 16);
      }
#pragma unroll
      for (int t = 0; t < 4; ++t) {
        acc[t].v = wmma_bf16(ah[t].v, bhf.v, acc[t].v);
        acc[t].v = wmma_bf16(ah[t].v, blf.v, acc[t].v);
        acc[t].v = wmma_bf16(al[t].v, bhf.v, acc[t].v);
      }
    }
  }

#pragma unroll
  for (int t = 0; t < 4; ++t)
#pragma unroll
    for (int r = 0; r < 8; ++r) {
      int m = m0 + t * 16 + r + 8 * lhalf;
      C[(size_t)m * N + n0 + lrow] = acc[t].e[r];
    }
}

// Split fp32 -> bf16 hi/lo pair (x ~= hi + lo).
__global__ void split_bf16_kernel(const float* __restrict__ src,
                                  bf16_t* __restrict__ hi,
                                  bf16_t* __restrict__ lo, size_t n) {
  size_t idx = (size_t)blockIdx.x * 256 + threadIdx.x;
  if (idx >= n) return;
  float v = src[idx];
  bf16_t h = (bf16_t)v;
  hi[idx] = h;
  lo[idx] = (bf16_t)(v - (float)h);
}

// Depthwise causal conv (K=4) + SiLU over the xBC channels of proj.
__global__ void conv_silu_kernel(const float* __restrict__ proj,
                                 const float* __restrict__ cw,
                                 const float* __restrict__ cb,
                                 float* __restrict__ xbc) {
  size_t idx = (size_t)blockIdx.x * 256 + threadIdx.x;
  const size_t total = (size_t)BATCH * LSEQ * DXBC;
  if (idx >= total) return;
  int c = (int)(idx % DXBC);
  int l = (int)((idx / DXBC) % LSEQ);
  int b = (int)(idx / ((size_t)DXBC * LSEQ));
  float acc = cb[c];
#pragma unroll
  for (int j = 0; j < CONVK; ++j) {
    int t = l - (CONVK - 1) + j;
    if (t >= 0)
      acc += cw[c * CONVK + j] * proj[((size_t)b * LSEQ + t) * DPROJ + c];
  }
  xbc[idx] = acc * sigmoidf(acc);
}

// dt[b,l,h] = softplus(proj[..., DXBC + DINNER + h])
__global__ void dt_kernel(const float* __restrict__ proj,
                          float* __restrict__ dt) {
  size_t idx = (size_t)blockIdx.x * 256 + threadIdx.x;
  const size_t total = (size_t)BATCH * LSEQ * NHEADS;
  if (idx >= total) return;
  int h = (int)(idx % NHEADS);
  size_t row = idx / NHEADS;
  float x = proj[row * DPROJ + DXBC + DINNER + h];
  dt[idx] = (x > 20.0f) ? x : log1pf(__expf(x));
}

// dacs[b,h,c,j] = cumsum_{i<=j} (-dt[b, c*CHUNK+i, h])
__global__ void dacs_kernel(const float* __restrict__ dt,
                            float* __restrict__ dacs) {
  int idx = blockIdx.x * 256 + threadIdx.x;
  const int total = BATCH * NHEADS * NCHUNK;
  if (idx >= total) return;
  int c = idx % NCHUNK;
  int h = (idx / NCHUNK) % NHEADS;
  int b = idx / (NCHUNK * NHEADS);
  float s = 0.0f;
  float* out = &dacs[((size_t)(b * NHEADS + h) * NCHUNK + c) * CHUNK];
  const float* src = &dt[((size_t)b * LSEQ + c * CHUNK) * NHEADS + h];
  for (int i = 0; i < CHUNK; ++i) {
    s -= src[(size_t)i * NHEADS];
    out[i] = s;
  }
}

// ---------------------------------------------------------------------------
// Per (b,chunk,head): G = C.B^T (128x128,K=64); Gm = G (.) L(decay mask);
// Y_diag = Gm.x (128x64,K=128) -> y; states = x^T.(decay (.) B) (64x64,K=128).
// ---------------------------------------------------------------------------
__global__ __launch_bounds__(256) void ssd_chunk_kernel(
    const float* __restrict__ xbc, const float* __restrict__ dacs,
    float* __restrict__ y, float* __restrict__ states) {
  __shared__ bf16_t Bs[CHUNK][DSTATE];    // [l][n] 16 KB
  __shared__ bf16_t xT[HEADDIM][CHUNK];   // [p][l] 16 KB
  __shared__ bf16_t Gm[CHUNK][CHUNK];     // [l][s] 32 KB

  const int c = blockIdx.x, h = blockIdx.y, b = blockIdx.z;
  const int wave = threadIdx.x >> 5;
  const int lane = threadIdx.x & 31;
  const int lrow = lane & 15, lhalf = lane >> 4;

  const size_t xbase = ((size_t)b * LSEQ + (size_t)c * CHUNK) * DXBC;
  const int xoff = h * HEADDIM;
  const int boff = DINNER + h * DSTATE;
  const int coff = DINNER + NHEADS * DSTATE + h * DSTATE;
  const float* dac = &dacs[((size_t)(b * NHEADS + h) * NCHUNK + c) * CHUNK];

  for (int i = threadIdx.x; i < CHUNK * DSTATE; i += 256) {
    int l = i >> 6, n = i & 63;
    Bs[l][n] = (bf16_t)xbc[xbase + (size_t)l * DXBC + boff + n];
    xT[n][l] = (bf16_t)xbc[xbase + (size_t)l * DXBC + xoff + n];
  }
  __syncthreads();

  // ---- GEMM1: G = C.B^T. Wave w owns rows 16w..16w+15, loops 8 s-blocks.
  FragAB aC[2];
#pragma unroll
  for (int ks = 0; ks < 2; ++ks) {
    const float* crow =
        &xbc[xbase + (size_t)(wave * 16 + lrow) * DXBC + coff + ks * 32];
#pragma unroll
    for (int e = 0; e < 16; ++e) aC[ks].e[e] = (bf16_t)crow[kb_idx(e, lhalf)];
  }
  float cum_row[8];
#pragma unroll
  for (int r = 0; r < 8; ++r) cum_row[r] = dac[wave * 16 + r + 8 * lhalf];

  for (int sb = 0; sb < 8; ++sb) {
    FragCD g;
#pragma unroll
    for (int i = 0; i < 8; ++i) g.e[i] = 0.0f;
#pragma unroll
    for (int ks = 0; ks < 2; ++ks) {
      FragAB bb;
      const bf16_t* brow = &Bs[sb * 16 + lrow][ks * 32 + 16 * lhalf];
      bb.q[0] = ((const uint4*)brow)[0];
      bb.q[1] = ((const uint4*)brow)[1];
      g.v = wmma_bf16(aC[ks].v, bb.v, g.v);
    }
    float cum_s = dac[sb * 16 + lrow];
    int s = sb * 16 + lrow;
#pragma unroll
    for (int r = 0; r < 8; ++r) {
      int l = wave * 16 + r + 8 * lhalf;
      float val = (s <= l) ? g.e[r] * __expf(cum_row[r] - cum_s) : 0.0f;
      Gm[l][s] = (bf16_t)val;
    }
  }
  __syncthreads();

  // ---- GEMM2: Y_diag = Gm.x. Wave w rows 16w.., 4 p-blocks, K=128.
  FragCD yacc[4];
#pragma unroll
  for (int t = 0; t < 4; ++t)
#pragma unroll
    for (int i = 0; i < 8; ++i) yacc[t].e[i] = 0.0f;

  for (int ks = 0; ks < 4; ++ks) {
    FragAB ag;
    const bf16_t* grow = &Gm[wave * 16 + lrow][ks * 32 + 16 * lhalf];
    ag.q[0] = *(const uint4*)grow;
    ag.q[1] = *(const uint4*)(grow + 8);
#pragma unroll
    for (int pb = 0; pb < 4; ++pb) {
      FragAB bx;
      const bf16_t* xr = &xT[pb * 16 + lrow][ks * 32 + 16 * lhalf];
      bx.q[0] = ((const uint4*)xr)[0];
      bx.q[1] = ((const uint4*)xr)[1];
      yacc[pb].v = wmma_bf16(ag.v, bx.v, yacc[pb].v);
    }
  }
#pragma unroll
  for (int pb = 0; pb < 4; ++pb)
#pragma unroll
    for (int r = 0; r < 8; ++r) {
      int l = wave * 16 + r + 8 * lhalf;
      int p = pb * 16 + lrow;
      y[((size_t)b * LSEQ + (size_t)c * CHUNK + l) * DINNER + h * HEADDIM + p] =
          yacc[pb].e[r];
    }

  // ---- states[p,n] = sum_l x[l,p] * exp(cum[last]-cum[l]) * B[l,n]
  float cumlast = dac[CHUNK - 1];
  for (int t = 2 * wave; t < 2 * wave + 2; ++t) {
    int pb = t >> 2, nb = t & 3;
    FragCD st;
#pragma unroll
    for (int i = 0; i < 8; ++i) st.e[i] = 0.0f;
    for (int ks = 0; ks < 4; ++ks) {
      FragAB ax;
      const bf16_t* xr = &xT[pb * 16 + lrow][ks * 32 + 8 * lhalf];
      ax.q[0] = *(const uint4*)xr;
      ax.q[1] = *(const uint4*)(xr + 16);
      FragAB bd;
#pragma unroll
      for (int e = 0; e < 16; ++e) {
        int kl = ks * 32 + ka_idx(e, lhalf);
        float bv = (float)Bs[kl][nb * 16 + lrow];
        bd.e[e] = (bf16_t)(bv * __expf(cumlast - dac[kl]));
      }
      st.v = wmma_bf16(ax.v, bd.v, st.v);
    }
#pragma unroll
    for (int r = 0; r < 8; ++r) {
      int p = pb * 16 + r + 8 * lhalf;
      int n = nb * 16 + lrow;
      states[(((size_t)((b * NCHUNK + c) * NHEADS + h)) * HEADDIM + p) *
                 DSTATE + n] = st.e[r];
    }
  }
}

// Inter-chunk scan: prev_c = carry; carry = carry*exp(dA_cum[c,last]) + s_c.
__global__ void chunk_scan_kernel(const float* __restrict__ states,
                                  const float* __restrict__ dacs,
                                  float* __restrict__ prev) {
  size_t idx = (size_t)blockIdx.x * 256 + threadIdx.x;
  const size_t total = (size_t)BATCH * NHEADS * HEADDIM * DSTATE;
  if (idx >= total) return;
  int n = (int)(idx & 63);
  int p = (int)((idx >> 6) & 63);
  int h = (int)((idx >> 12) & 31);
  int b = (int)(idx >> 17);
  float carry = 0.0f;
  for (int c = 0; c < NCHUNK; ++c) {
    size_t sidx =
        (((size_t)((b * NCHUNK + c) * NHEADS + h)) * HEADDIM + p) * DSTATE + n;
    prev[sidx] = carry;
    float dec = __expf(
        dacs[((size_t)(b * NHEADS + h) * NCHUNK + c) * CHUNK + CHUNK - 1]);
    carry = carry * dec + states[sidx];
  }
}

// Per (b,chunk,head): y += exp(dA_cum[l]) * (C.prev^T) + D[h]*x.
__global__ __launch_bounds__(256) void ssd_off_kernel(
    const float* __restrict__ xbc, const float* __restrict__ dacs,
    const float* __restrict__ prev, const float* __restrict__ Dvec,
    float* __restrict__ y) {
  __shared__ bf16_t Pv[HEADDIM][DSTATE];  // [p][n] 8 KB
  __shared__ float dac_s[CHUNK];

  const int c = blockIdx.x, h = blockIdx.y, b = blockIdx.z;
  const int wave = threadIdx.x >> 5;
  const int lane = threadIdx.x & 31;
  const int lrow = lane & 15, lhalf = lane >> 4;

  const size_t xbase = ((size_t)b * LSEQ + (size_t)c * CHUNK) * DXBC;
  const int xoff = h * HEADDIM;
  const int coff = DINNER + NHEADS * DSTATE + h * DSTATE;
  const float* dac = &dacs[((size_t)(b * NHEADS + h) * NCHUNK + c) * CHUNK];
  const size_t pbase =
      ((size_t)((b * NCHUNK + c) * NHEADS + h)) * HEADDIM * DSTATE;

  for (int i = threadIdx.x; i < HEADDIM * DSTATE; i += 256)
    Pv[i >> 6][i & 63] = (bf16_t)prev[pbase + i];
  if (threadIdx.x < CHUNK) dac_s[threadIdx.x] = dac[threadIdx.x];
  __syncthreads();

  FragAB aC[2];
#pragma unroll
  for (int ks = 0; ks < 2; ++ks) {
    const float* crow =
        &xbc[xbase + (size_t)(wave * 16 + lrow) * DXBC + coff + ks * 32];
#pragma unroll
    for (int e = 0; e < 16; ++e) aC[ks].e[e] = (bf16_t)crow[kb_idx(e, lhalf)];
  }

  FragCD acc[4];
#pragma unroll
  for (int t = 0; t < 4; ++t)
#pragma unroll
    for (int i = 0; i < 8; ++i) acc[t].e[i] = 0.0f;

#pragma unroll
  for (int ks = 0; ks < 2; ++ks)
#pragma unroll
    for (int pb = 0; pb < 4; ++pb) {
      FragAB bp;
      const bf16_t* pr = &Pv[pb * 16 + lrow][ks * 32 + 16 * lhalf];
      bp.q[0] = ((const uint4*)pr)[0];
      bp.q[1] = ((const uint4*)pr)[1];
      acc[pb].v = wmma_bf16(aC[ks].v, bp.v, acc[pb].v);
    }

  float Dh = Dvec[h];
#pragma unroll
  for (int pb = 0; pb < 4; ++pb)
#pragma unroll
    for (int r = 0; r < 8; ++r) {
      int l = wave * 16 + r + 8 * lhalf;
      int p = pb * 16 + lrow;
      size_t yi =
          ((size_t)b * LSEQ + (size_t)c * CHUNK + l) * DINNER + h * HEADDIM + p;
      float xv = xbc[xbase + (size_t)l * DXBC + xoff + p];
      y[yi] += acc[pb].e[r] * __expf(dac_s[l]) + Dh * xv;
    }
}

// Gating fused with bf16 hi/lo split: yg = y * silu(z + z_bias) -> (hi, lo)
__global__ void gate_split_kernel(const float* __restrict__ proj,
                                  const float* __restrict__ zb,
                                  const float* __restrict__ y,
                                  bf16_t* __restrict__ yhi,
                                  bf16_t* __restrict__ ylo) {
  size_t idx = (size_t)blockIdx.x * 256 + threadIdx.x;
  const size_t total = (size_t)BATCH * LSEQ * DINNER;
  if (idx >= total) return;
  int d = (int)(idx % DINNER);
  size_t row = idx / DINNER;
  float zv = proj[row * DPROJ + DXBC + d] + zb[d];
  float g = y[idx] * (zv * sigmoidf(zv));
  bf16_t hh = (bf16_t)g;
  yhi[idx] = hh;
  ylo[idx] = (bf16_t)(g - (float)hh);
}

// ---------------------------------------------------------------------------
extern "C" void kernel_launch(void* const* d_in, const int* in_sizes, int n_in,
                              void* d_out, int out_size, void* d_ws,
                              size_t ws_size, hipStream_t stream) {
  const float* u          = (const float*)d_in[0];
  const float* in_proj_w  = (const float*)d_in[1];
  const float* z_bias     = (const float*)d_in[2];
  const float* conv_w     = (const float*)d_in[3];
  const float* conv_b     = (const float*)d_in[4];
  const float* Dvec       = (const float*)d_in[5];
  const float* out_proj_w = (const float*)d_in[6];
  float* out = (float*)d_out;

  const size_t n_u  = (size_t)BATCH * LSEQ * DMODEL;   // 8,388,608
  const size_t n_w1 = (size_t)DPROJ * DMODEL;          // 8,421,376
  const size_t n_w2 = (size_t)DMODEL * DINNER;         // 2,097,152
  const size_t n_y  = (size_t)BATCH * LSEQ * DINNER;   // 16,777,216

  char* ws = (char*)d_ws;
  float* proj = (float*)ws;  ws += sizeof(float) * (size_t)BATCH * LSEQ * DPROJ;
  float* xbc  = (float*)ws;  ws += sizeof(float) * (size_t)BATCH * LSEQ * DXBC;
  float* dt   = (float*)ws;  ws += sizeof(float) * (size_t)BATCH * LSEQ * NHEADS;
  float* dacs = (float*)ws;  ws += sizeof(float) * (size_t)BATCH * NHEADS * LSEQ;
  float* states = (float*)ws;
  ws += sizeof(float) * (size_t)BATCH * NCHUNK * NHEADS * HEADDIM * DSTATE;
  float* prev = (float*)ws;
  ws += sizeof(float) * (size_t)BATCH * NCHUNK * NHEADS * HEADDIM * DSTATE;
  float* y = (float*)ws;     ws += sizeof(float) * n_y;
  bf16_t* uhi  = (bf16_t*)ws; ws += sizeof(bf16_t) * n_u;
  bf16_t* ulo  = (bf16_t*)ws; ws += sizeof(bf16_t) * n_u;
  bf16_t* w1hi = (bf16_t*)ws; ws += sizeof(bf16_t) * n_w1;
  bf16_t* w1lo = (bf16_t*)ws; ws += sizeof(bf16_t) * n_w1;
  bf16_t* w2hi = (bf16_t*)ws; ws += sizeof(bf16_t) * n_w2;
  bf16_t* w2lo = (bf16_t*)ws; ws += sizeof(bf16_t) * n_w2;
  bf16_t* yhi  = (bf16_t*)ws; ws += sizeof(bf16_t) * n_y;
  bf16_t* ylo  = (bf16_t*)ws; ws += sizeof(bf16_t) * n_y;

  const int M = BATCH * LSEQ;  // 8192

  // 0) precision-split pre-passes (bandwidth-trivial)
  split_bf16_kernel<<<(int)((n_u + 255) / 256), 256, 0, stream>>>(u, uhi, ulo,
                                                                  n_u);
  split_bf16_kernel<<<(int)((n_w1 + 255) / 256), 256, 0, stream>>>(
      in_proj_w, w1hi, w1lo, n_w1);
  split_bf16_kernel<<<(int)((n_w2 + 255) / 256), 256, 0, stream>>>(
      out_proj_w, w2hi, w2lo, n_w2);

  // 1) proj = u @ in_proj_w^T   (M x 8224, K = 1024)
  {
    dim3 grid(M / 64, (DPROJ / 16 + 7) / 8);
    gemm_nt_bf16x2<<<grid, 256, 0, stream>>>(uhi, ulo, w1hi, w1lo, proj, M,
                                             DPROJ, DMODEL);
  }
  // 2) depthwise conv + SiLU
  {
    size_t total = (size_t)BATCH * LSEQ * DXBC;
    conv_silu_kernel<<<(int)((total + 255) / 256), 256, 0, stream>>>(
        proj, conv_w, conv_b, xbc);
  }
  // 3) dt = softplus(A_log)
  {
    size_t total = (size_t)BATCH * LSEQ * NHEADS;
    dt_kernel<<<(int)((total + 255) / 256), 256, 0, stream>>>(proj, dt);
  }
  // 4) per-chunk cumsum of dA = -dt
  {
    int total = BATCH * NHEADS * NCHUNK;
    dacs_kernel<<<(total + 255) / 256, 256, 0, stream>>>(dt, dacs);
  }
  // 5) intra-chunk SSD (Y_diag + states)
  {
    dim3 grid(NCHUNK, NHEADS, BATCH);
    ssd_chunk_kernel<<<grid, 256, 0, stream>>>(xbc, dacs, y, states);
  }
  // 6) inter-chunk scan
  {
    size_t total = (size_t)BATCH * NHEADS * HEADDIM * DSTATE;
    chunk_scan_kernel<<<(int)((total + 255) / 256), 256, 0, stream>>>(
        states, dacs, prev);
  }
  // 7) Y_off + D*x
  {
    dim3 grid(NCHUNK, NHEADS, BATCH);
    ssd_off_kernel<<<grid, 256, 0, stream>>>(xbc, dacs, prev, Dvec, y);
  }
  // 8) gating + split: (yhi, ylo) = split(y * silu(z + z_bias))
  gate_split_kernel<<<(int)((n_y + 255) / 256), 256, 0, stream>>>(
      proj, z_bias, y, yhi, ylo);
  // 9) out = yg @ out_proj_w^T   (M x 1024, K = 2048)
  {
    dim3 grid(M / 64, (DMODEL / 16 + 7) / 8);
    gemm_nt_bf16x2<<<grid, 256, 0, stream>>>(yhi, ylo, w2hi, w2lo, out, M,
                                             DMODEL, DINNER);
  }
}
